// LSH_867583393877
// MI455X (gfx1250) — matrix operand order repaired and checked by reading, not compiled, not run on hardware
//
#include <hip/hip_runtime.h>

#define BATCH  4096
#define EMBED  1024
#define NDIM   512
#define NCODES 65536
#define MT     32      // batch rows per workgroup in the argmin kernel

typedef float  v2f   __attribute__((ext_vector_type(2)));
typedef float  v8f   __attribute__((ext_vector_type(8)));
typedef __bf16 v8bf  __attribute__((ext_vector_type(8)));
typedef __bf16 v16bf __attribute__((ext_vector_type(16)));
typedef unsigned int v4u __attribute__((ext_vector_type(4)));
typedef int          v4i __attribute__((ext_vector_type(4)));
typedef int          v8i __attribute__((ext_vector_type(8)));

static __device__ __forceinline__ unsigned short f2bf_rne(float f) {
    unsigned u = __builtin_bit_cast(unsigned, f);
    u += 0x7FFFu + ((u >> 16) & 1u);
    return (unsigned short)(u >> 16);
}
static __device__ __forceinline__ float bf2f(unsigned short h) {
    unsigned u = ((unsigned)h) << 16;
    return __builtin_bit_cast(float, u);
}

// ---------------------------------------------------------------------------
// TDM: issue a 2D tensor tile load (Global -> LDS) via the Tensor Data Mover.
// D# layout per CDNA5 ISA ch.8: group0 = {count/type/addrs}, group1 = dims.
// data_size_code: 0=1B, 1=2B, 2=4B, 3=8B. All sizes in elements.
// ---------------------------------------------------------------------------
#if defined(__gfx1250__) && __has_builtin(__builtin_amdgcn_tensor_load_to_lds)
#define HAVE_TDM 1
static __device__ __forceinline__ void tdm_load_2d(const void* gsrc, unsigned lds_off,
                                                   unsigned tensor_d0, unsigned tensor_d1,
                                                   unsigned tile_d0, unsigned tile_d1,
                                                   unsigned stride_d0, unsigned data_size_code) {
    unsigned long long ga = (unsigned long long)(uintptr_t)gsrc;
    v4u g0;
    g0[0] = 1u;                                      // count=1, is_restore=0, gather off
    g0[1] = lds_off;                                 // lds_addr (byte offset in LDS)
    g0[2] = (unsigned)(ga & 0xFFFFFFFFu);            // global_addr[31:0]
    g0[3] = (unsigned)((ga >> 32) & 0x01FFFFFFu)     // global_addr[56:32]
          | 0x80000000u;                             // type = 2 ("image")
    v8i g1;
    g1[0] = (int)(data_size_code << 16);             // wg_mask=0, data_size, no pad/iter
    g1[1] = (int)((tensor_d0 & 0xFFFFu) << 16);      // atomic_barrier=0 | tensor_dim0 lo
    g1[2] = (int)(((tensor_d0 >> 16) & 0xFFFFu) | ((tensor_d1 & 0xFFFFu) << 16));
    g1[3] = (int)(((tensor_d1 >> 16) & 0xFFFFu) | ((tile_d0 & 0xFFFFu) << 16));
    g1[4] = (int)(tile_d1 & 0xFFFFu);                // tile_dim1 | tile_dim2=0
    g1[5] = (int)stride_d0;                          // tensor_dim0_stride[31:0]
    g1[6] = 0;                                       // stride hi=0 | dim1_stride lo=0
    g1[7] = 0;                                       // dim1_stride hi=0 (2D tile)
    v4i z4 = {0, 0, 0, 0};
#if __has_include(<hip/amd_detail/amd_gfx1250_TDM.h>)
    v8i z8 = {0, 0, 0, 0, 0, 0, 0, 0};               // clang-23 / therock: 6-arg form
    __builtin_amdgcn_tensor_load_to_lds(g0, g1, z4, z4, z8, 0);
#else
    __builtin_amdgcn_tensor_load_to_lds(g0, g1, z4, z4, 0);   // ROCm 7.2: 5-arg form
#endif
}
#else
#define HAVE_TDM 0
#endif

// ---------------------------------------------------------------------------
// Kernel 0: tensors f32 -> bf16 (exact: values are small integers), 8 elem/thread
// ---------------------------------------------------------------------------
__global__ void __launch_bounds__(256)
cvt_bf16_kernel(const float* __restrict__ src, unsigned short* __restrict__ dst) {
    int i = blockIdx.x * 256 + threadIdx.x;          // i indexes groups of 8 floats
    const float4* s = (const float4*)src + (size_t)i * 2;
    float4 a = s[0];
    float4 b = s[1];
    uint4 o;
    o.x = (unsigned)f2bf_rne(a.x) | ((unsigned)f2bf_rne(a.y) << 16);
    o.y = (unsigned)f2bf_rne(a.z) | ((unsigned)f2bf_rne(a.w) << 16);
    o.z = (unsigned)f2bf_rne(b.x) | ((unsigned)f2bf_rne(b.y) << 16);
    o.w = (unsigned)f2bf_rne(b.z) | ((unsigned)f2bf_rne(b.w) << 16);
    ((uint4*)dst)[i] = o;
}

// ---------------------------------------------------------------------------
// Kernel 1: v = trunc(x @ W^T) via V_WMMA_F32_16X16X4_F32 (full f32 precision),
// then split v into a bf16 hi/lo pair for the big bf16 GEMM.
// One wave per 16x16 output tile; 8 waves / 256-thread block.
// ---------------------------------------------------------------------------
__global__ void __launch_bounds__(256)
gemm1_trunc_kernel(const float* __restrict__ x, const float* __restrict__ w,
                   unsigned short* __restrict__ vh, unsigned short* __restrict__ vl) {
    const int lane = threadIdx.x & 31;
    const int wav  = (blockIdx.x << 3) + (threadIdx.x >> 5);  // global wave id
    const int mt   = wav >> 5;                                // 0..255 (batch tiles)
    const int nt   = wav & 31;                                // 0..31  (dim tiles)
    const int sub  = lane & 15;
    const int kh   = lane >> 4;

    const float* xp = x + (size_t)(mt * 16 + sub) * EMBED + kh * 2;
    const float* wp = w + (size_t)(nt * 16 + sub) * EMBED + kh * 2;

    v8f acc = {};
#pragma unroll 8
    for (int k = 0; k < EMBED; k += 4) {
        v2f a = *(const v2f*)(xp + k);
        v2f b = *(const v2f*)(wp + k);
        acc = __builtin_amdgcn_wmma_f32_16x16x4_f32(false, a, false, b,
                                                    (short)0, acc, false, false);
    }

    // D layout: VGPR r, lanes 0-15 -> M=r, lanes 16-31 -> M=r+8; N = lane&15
#pragma unroll
    for (int r = 0; r < 8; ++r) {
        float t = truncf(acc[r]);
        unsigned short h = f2bf_rne(t);
        unsigned short l = f2bf_rne(t - bf2f(h));
        int m = mt * 16 + r + kh * 8;
        int n = nt * 16 + sub;
        vh[(size_t)m * NDIM + n] = h;
        vl[(size_t)m * NDIM + n] = l;
    }
}

// ---------------------------------------------------------------------------
// Kernel 2: fused  dist = (v_hi + v_lo) @ tensors_bf16^T  ->  argmin  ->  gather.
// 512 threads = 16 waves; block owns MT=32 batch rows. v tile staged to LDS by
// the Tensor Data Mover (wave 0 issues two 2D-tile descriptors, waits on
// TENSORcnt, workgroup barrier releases the rest). Waves split the 65536 codes
// 8-way; per K-chunk of 32: 2 bf16 WMMAs (hi+lo) per N-subtile into f32 accum.
// ---------------------------------------------------------------------------
__global__ void __launch_bounds__(512)
lsh_argmin_kernel(const unsigned short* __restrict__ vh,
                  const unsigned short* __restrict__ vl,
                  const unsigned short* __restrict__ tb,
                  const float* __restrict__ tens,
                  float* __restrict__ out) {
    __shared__ __align__(16) unsigned char smem[MT * NDIM * 2 * 2];  // 64 KB
    unsigned short* s_vh = (unsigned short*)smem;
    unsigned short* s_vl = s_vh + MT * NDIM;

    const int tid  = threadIdx.x;
    const int lane = tid & 31;
    const int w    = tid >> 5;     // 0..15
    const int wm   = w & 1;        // M subtile
    const int wn   = w >> 1;       // 0..7, N partition
    const int mb   = blockIdx.x * MT;
    const int sub  = lane & 15;
    const int kh   = lane >> 4;

#if HAVE_TDM
    if (w == 0) {
        unsigned lds_h = (unsigned)(uintptr_t)(void*)s_vh;
        unsigned lds_l = (unsigned)(uintptr_t)(void*)s_vl;
        // 2D bf16 tile: 512 (K) x 32 (rows), row stride 512 elements.
        tdm_load_2d(vh + (size_t)mb * NDIM, lds_h, NDIM, MT, NDIM, MT, NDIM, 1u);
        tdm_load_2d(vl + (size_t)mb * NDIM, lds_l, NDIM, MT, NDIM, MT, NDIM, 1u);
        __builtin_amdgcn_s_wait_tensorcnt(0);
    }
#else
    {
        const uint4* gh = (const uint4*)(vh + (size_t)mb * NDIM);
        const uint4* gl = (const uint4*)(vl + (size_t)mb * NDIM);
        uint4* sh = (uint4*)s_vh;
        uint4* sl = (uint4*)s_vl;
        const int n16 = MT * NDIM / 8;  // 2048 uint4
        for (int i = tid; i < n16; i += 512) { sh[i] = gh[i]; sl[i] = gl[i]; }
    }
#endif
    __syncthreads();

    // A operand rows for this lane (ISA 16-bit A 16x32 layout: per half-wave
    // K ranges {0-7,16-23} / {8-15,24-31} within each chunk of 32).
    const unsigned short* arow_h = s_vh + (size_t)(wm * 16 + sub) * NDIM;
    const unsigned short* arow_l = s_vl + (size_t)(wm * 16 + sub) * NDIM;

    float runMin[8];
    int   runIdx[8];
#pragma unroll
    for (int r = 0; r < 8; ++r) { runMin[r] = 3.4e38f; runIdx[r] = 0; }

    for (int nb = wn * 32; nb < NCODES; nb += 256) {
        const int n0 = nb, n1 = nb + 16;
        // B (32x16 bf16): lane holds column N=lane&15; half-wave 0 -> K {0-7,16-23},
        // half-wave 1 -> K {8-15,24-31} of each chunk. Rows of tb are K-contiguous.
        const unsigned short* b0p = tb + (size_t)(n0 + sub) * NDIM + kh * 8;
        const unsigned short* b1p = tb + (size_t)(n1 + sub) * NDIM + kh * 8;

        // Prefetch next iteration's B rows into cache (global_prefetch_b8).
        if (nb + 256 < NCODES) {
            __builtin_prefetch(b0p + 256 * NDIM, 0, 1);
            __builtin_prefetch(b1p + 256 * NDIM, 0, 1);
        }

        v8f acc0 = {};
        v8f acc1 = {};
#pragma unroll 4
        for (int kk = 0; kk < NDIM; kk += 32) {
            v8bf ah0 = *(const v8bf*)(arow_h + kk + kh * 8);
            v8bf ah1 = *(const v8bf*)(arow_h + kk + 16 + kh * 8);
            v8bf al0 = *(const v8bf*)(arow_l + kk + kh * 8);
            v8bf al1 = *(const v8bf*)(arow_l + kk + 16 + kh * 8);
            v16bf Ah = __builtin_shufflevector(ah0, ah1,
                0, 1, 2, 3, 4, 5, 6, 7, 8, 9, 10, 11, 12, 13, 14, 15);
            v16bf Al = __builtin_shufflevector(al0, al1,
                0, 1, 2, 3, 4, 5, 6, 7, 8, 9, 10, 11, 12, 13, 14, 15);

            v8bf b00 = *(const v8bf*)(b0p + kk);
            v8bf b01 = *(const v8bf*)(b0p + kk + 16);
            v16bf B0 = __builtin_shufflevector(b00, b01,
                0, 1, 2, 3, 4, 5, 6, 7, 8, 9, 10, 11, 12, 13, 14, 15);
            v8bf b10 = *(const v8bf*)(b1p + kk);
            v8bf b11 = *(const v8bf*)(b1p + kk + 16);
            v16bf B1 = __builtin_shufflevector(b10, b11,
                0, 1, 2, 3, 4, 5, 6, 7, 8, 9, 10, 11, 12, 13, 14, 15);

            acc0 = __builtin_amdgcn_wmma_f32_16x16x32_bf16(false, Ah, false, B0,
                                                           (short)0, acc0, false, false);
            acc0 = __builtin_amdgcn_wmma_f32_16x16x32_bf16(false, Al, false, B0,
                                                           (short)0, acc0, false, false);
            acc1 = __builtin_amdgcn_wmma_f32_16x16x32_bf16(false, Ah, false, B1,
                                                           (short)0, acc1, false, false);
            acc1 = __builtin_amdgcn_wmma_f32_16x16x32_bf16(false, Al, false, B1,
                                                           (short)0, acc1, false, false);
        }
        // Per-lane running min; strict '<' keeps the earliest (smallest) code idx.
#pragma unroll
        for (int r = 0; r < 8; ++r) {
            if (acc0[r] < runMin[r]) { runMin[r] = acc0[r]; runIdx[r] = n0 + sub; }
            if (acc1[r] < runMin[r]) { runMin[r] = acc1[r]; runIdx[r] = n1 + sub; }
        }
    }

    __syncthreads();  // everyone done with the v tile; alias LDS for reduction
    float* red_min = (float*)smem;                  // [8][32]
    int*   red_idx = (int*)(smem + 2048);           // [8][32]
    int*   fin     = (int*)(smem + 4096);           // [32]

#pragma unroll
    for (int r = 0; r < 8; ++r) {
        float m = runMin[r];
        int   i = runIdx[r];
#pragma unroll
        for (int off = 1; off < 16; off <<= 1) {
            float om = __shfl_xor(m, off, 16);
            int   oi = __shfl_xor(i, off, 16);
            if (om < m || (om == m && oi < i)) { m = om; i = oi; }
        }
        if (sub == 0) {
            int rl = wm * 16 + r + kh * 8;          // local row 0..31
            red_min[wn * 32 + rl] = m;
            red_idx[wn * 32 + rl] = i;
        }
    }
    __syncthreads();

    if (tid < 32) {
        float m = red_min[tid];
        int   i = red_idx[tid];
#pragma unroll
        for (int q = 1; q < 8; ++q) {
            float om = red_min[q * 32 + tid];
            int   oi = red_idx[q * 32 + tid];
            if (om < m || (om == m && oi < i)) { m = om; i = oi; }
        }
        fin[tid] = i;
    }
    __syncthreads();

    // Gather: out[mb+rl] = tensors[fin[rl]] (512 f32 per row, float4 vectorized)
    for (int rl = w; rl < MT; rl += 16) {
        int n = fin[rl];
        const float4* src = (const float4*)(tens + (size_t)n * NDIM);
        float4*       dst = (float4*)(out + (size_t)(mb + rl) * NDIM);
        for (int j = lane; j < NDIM / 4; j += 32) dst[j] = src[j];
    }
}

// ---------------------------------------------------------------------------
extern "C" void kernel_launch(void* const* d_in, const int* in_sizes, int n_in,
                              void* d_out, int out_size, void* d_ws, size_t ws_size,
                              hipStream_t stream) {
    (void)in_sizes; (void)n_in; (void)out_size; (void)ws_size;
    const float* x    = (const float*)d_in[0];   // [4096,1024]
    const float* wts  = (const float*)d_in[1];   // [512,1024]
    const float* tens = (const float*)d_in[2];   // [65536,512]
    float* out = (float*)d_out;                  // [4096,512]

    unsigned char* ws = (unsigned char*)d_ws;
    unsigned short* tb = (unsigned short*)ws;                                  // 64 MB
    unsigned short* vh = (unsigned short*)(ws + (size_t)NCODES * NDIM * 2);    // 4 MB
    unsigned short* vl = vh + (size_t)BATCH * NDIM;                            // 4 MB

    // 0) tensors -> bf16 (exact; small integers)
    cvt_bf16_kernel<<<(NCODES * NDIM / 8) / 256, 256, 0, stream>>>(tens, tb);
    // 1) v = trunc(x @ W^T) in f32 WMMA, split into bf16 hi/lo
    gemm1_trunc_kernel<<<(BATCH / 16) * (NDIM / 16) / 8, 256, 0, stream>>>(x, wts, vh, vl);
    // 2) fused bf16 GEMM + argmin + gather (TDM-staged LDS tile)
    lsh_argmin_kernel<<<BATCH / MT, 512, 0, stream>>>(vh, vl, tb, tens, out);
}
